// GroupedQueryAttention_73993696576148
// MI455X (gfx1250) — matrix operand order repaired and checked by reading, not compiled, not run on hardware
//
#include <hip/hip_runtime.h>
#include <hip/hip_bf16.h>
#include <math.h>

// ---------------------------------------------------------------------------
// GQA forward on gfx1250 (wave32, WMMA f32_16x16x32_f16).
//   proj:  Y = (x @ W + b) * outScale -> f16 (V stored transposed [B,NG*64,S])
//          Q is pre-scaled by (1/sqrt(64))*log2(e) so attention runs in the
//          exp2 domain (raw v_exp_f32 via __builtin_amdgcn_exp2f).
//   attn:  flash-style online softmax; row-sums accumulated with an extra
//          WMMA against an all-ones B tile (no cross-lane sum reductions);
//          shared (tile-wide) running max -> single width-32 shuffle tree.
//          K fragments batch-loaded before the score WMMAs; V fragments
//          batch-loaded before the softmax VALU so they are in flight while
//          the exp/max work executes.
// ---------------------------------------------------------------------------

typedef __attribute__((ext_vector_type(16))) _Float16 v16h;
typedef __attribute__((ext_vector_type(8)))  _Float16 v8h;
typedef __attribute__((ext_vector_type(8)))  float    v8f;

union Frag16 { v16h v; v8h h[2]; };

__device__ __forceinline__ v8f wmma_f16(const v16h a, const v16h b, const v8f c) {
    // 8 args: (neg_a, A, neg_b, B, c_mod, C, reuse_a, reuse_b)
    return __builtin_amdgcn_wmma_f32_16x16x32_f16(false, a, false, b, (short)0, c, false, false);
}

// Raw hardware exp2 (v_exp_f32): ~1ulp, flushes results below 2^-126 to zero,
// which is exactly the desired softmax-tail behavior (args are always <= 0).
__device__ __forceinline__ float fast_exp2(float x) {
    return __builtin_amdgcn_exp2f(x);
}

constexpr int B_   = 4;
constexpr int S_   = 2048;
constexpr int DIN_ = 1024;
constexpr int NH_  = 16;
constexpr int NG_  = 4;
constexpr int DK_  = 1024;   // NH*64
constexpr int DKV_ = 256;    // NG*64

// ---------------------------------------------------------------------------
// Projection GEMM: out[m][n] = (sum_k x[m][k]*W[k][n] + bias[n]) * outScale.
// Block = 128 threads (4 waves). Tile M=64 (16 rows/wave), N=128, K-step 32.
// 8 WMMAs per wave per k-step off one A fragment (good staging amortization).
// LDS: xs[64][32] f16; ws[128 n][32 k] f16 (W staged transposed so WMMA
// B-fragments are two contiguous ds_load_b128 per lane).
// ---------------------------------------------------------------------------
template <int NCOLS, bool TRANSPOSE_OUT>
__global__ __launch_bounds__(128)
void proj_kernel(const float* __restrict__ x, const float* __restrict__ W,
                 const float* __restrict__ bias, _Float16* __restrict__ out,
                 float outScale)
{
    __shared__ _Float16 xs[64][32];
    __shared__ _Float16 ws[128][32];

    const int tid  = threadIdx.x;
    const int wave = tid >> 5;
    const int lane = tid & 31;
    const int hi   = lane >> 4;
    const int ln   = lane & 15;

    const int m0 = blockIdx.x * 64;
    const int n0 = blockIdx.y * 128;

    const v8f zf = {0.f, 0.f, 0.f, 0.f, 0.f, 0.f, 0.f, 0.f};
    v8f acc[8] = {zf, zf, zf, zf, zf, zf, zf, zf};

    for (int k0 = 0; k0 < DIN_; k0 += 32) {
        __syncthreads();

        // stage x tile: 64 rows x 32 k, f32 -> f16
        {
            const int row  = tid >> 1;
            const int kseg = (tid & 1) * 16;
            const float* xp = x + (size_t)(m0 + row) * DIN_ + k0 + kseg;
            alignas(16) _Float16 hbuf[16];
            #pragma unroll
            for (int i = 0; i < 16; ++i) hbuf[i] = (_Float16)xp[i];
            *(v8h*)&xs[row][kseg]     = *(const v8h*)&hbuf[0];
            *(v8h*)&xs[row][kseg + 8] = *(const v8h*)&hbuf[8];
        }
        // stage W tile transposed: ws[n][kk] (coalesced per-kk row loads)
        {
            const int col = tid;                      // 0..127
            alignas(16) _Float16 hbuf[32];
            #pragma unroll
            for (int i = 0; i < 32; ++i)
                hbuf[i] = (_Float16)W[(size_t)(k0 + i) * NCOLS + n0 + col];
            #pragma unroll
            for (int q = 0; q < 4; ++q)
                *(v8h*)&ws[col][q * 8] = *(const v8h*)&hbuf[q * 8];
        }
        __syncthreads();

        Frag16 af;
        const int arow = wave * 16 + ln;
        af.h[0] = *(const v8h*)&xs[arow][hi * 8];
        af.h[1] = *(const v8h*)&xs[arow][16 + hi * 8];

        #pragma unroll
        for (int t = 0; t < 8; ++t) {
            Frag16 bf;
            const int bcol = t * 16 + ln;
            bf.h[0] = *(const v8h*)&ws[bcol][hi * 8];
            bf.h[1] = *(const v8h*)&ws[bcol][16 + hi * 8];
            acc[t] = wmma_f16(af.v, bf.v, acc[t]);
        }
    }

    #pragma unroll
    for (int t = 0; t < 8; ++t) {
        const int ncol = n0 + t * 16 + ln;
        const float bv = bias[ncol];
        #pragma unroll
        for (int r = 0; r < 8; ++r) {
            const int mrow = m0 + wave * 16 + r + 8 * hi;
            const float val = (acc[t][r] + bv) * outScale;
            if (!TRANSPOSE_OUT) {
                out[(size_t)mrow * NCOLS + ncol] = (_Float16)val;
            } else {
                const int bb = mrow >> 11;        // / S_
                const int ss = mrow & (S_ - 1);   // % S_
                out[(size_t)bb * NCOLS * S_ + (size_t)ncol * S_ + ss] = (_Float16)val;
            }
        }
    }
}

// ---------------------------------------------------------------------------
// Flash attention. Block = 128 threads (4 waves); each wave owns 16 q-rows.
// grid = (S/64, B*NH). Keys in chunks of 32. Scores arrive already in the
// exp2 domain (Q pre-scaled by 0.125*log2e in the projection).
//   Qh: [B*S, NH*64] f16   Kh: [B*S, NG*64] f16   Vt: [B, NG*64, S] f16
//   out: [B*S, NH*64] f32
// ---------------------------------------------------------------------------
__global__ __launch_bounds__(128)
void attn_kernel(const _Float16* __restrict__ Qh, const _Float16* __restrict__ Kh,
                 const _Float16* __restrict__ Vt, float* __restrict__ outp)
{
    __shared__ _Float16 Pst[4][16][32];   // per-wave C-layout -> A-layout bounce

    const int tid  = threadIdx.x;
    const int wave = tid >> 5;
    const int lane = tid & 31;
    const int hi   = lane >> 4;
    const int ln   = lane & 15;

    const int bh = blockIdx.y;
    const int b  = bh / NH_;
    const int h  = bh % NH_;
    const int g  = h / (NH_ / NG_);
    const int m0 = blockIdx.x * 64 + wave * 16;

    // Q A-fragments (2 chunks of K=32 covering head dim 64); already scaled.
    Frag16 qf[2];
    {
        const int qrow = m0 + ln;
        const _Float16* qp = Qh + ((size_t)(b * S_ + qrow) * NH_ + h) * 64;
        #pragma unroll
        for (int c = 0; c < 2; ++c) {
            qf[c].h[0] = *(const v8h*)(qp + c * 32 + hi * 16);
            qf[c].h[1] = *(const v8h*)(qp + c * 32 + hi * 16 + 8);
        }
    }

    // all-ones 32x16 B fragment: row-sum accumulator via WMMA
    Frag16 ones;
    #pragma unroll
    for (int i = 0; i < 16; ++i) ones.v[i] = (_Float16)1.0f;

    const v8f zf = {0.f, 0.f, 0.f, 0.f, 0.f, 0.f, 0.f, 0.f};
    v8f o[4] = {zf, zf, zf, zf};
    v8f lacc = zf;               // per-row exp-sums, replicated across 16 lanes
    float mrun = -3.0e38f;       // shared (tile-wide) running max, exp2 domain

    const _Float16* kcol  = Kh + ((size_t)(b * S_ + ln) * NG_ + g) * 64;
    const _Float16* vbase = Vt + (size_t)b * DKV_ * S_ + (size_t)(g * 64) * S_;

    for (int j0 = 0; j0 < S_; j0 += 32) {
        // prefetch next chunk of K / V into cache (global_prefetch_b8)
        {
            const int jn = (j0 + 32 < S_) ? (j0 + 32) : j0;
            __builtin_prefetch((const void*)(kcol + (size_t)jn * DKV_), 0, 1);
            __builtin_prefetch((const void*)(vbase + (size_t)ln * S_ + jn), 0, 1);
        }

        // ---- batch-load all 4 K fragments (one clause, staggered waits) -----
        Frag16 kf[2][2];          // [tile(+0/+16 keys)][c(head-dim chunk)]
        {
            const _Float16* kp = kcol + (size_t)j0 * DKV_ + hi * 16;
            #pragma unroll
            for (int tt = 0; tt < 2; ++tt)
                #pragma unroll
                for (int c = 0; c < 2; ++c) {
                    const _Float16* p = kp + (size_t)(tt * 16) * DKV_ + c * 32;
                    kf[tt][c].h[0] = *(const v8h*)(p);
                    kf[tt][c].h[1] = *(const v8h*)(p + 8);
                }
        }

        // ---- scores: two 16x16 tiles (keys j0..+16, j0+16..+32) -------------
        v8f s0 = zf, s1 = zf;
        s0 = wmma_f16(qf[0].v, kf[0][0].v, s0);
        s0 = wmma_f16(qf[1].v, kf[0][1].v, s0);
        s1 = wmma_f16(qf[0].v, kf[1][0].v, s1);
        s1 = wmma_f16(qf[1].v, kf[1][1].v, s1);

        // ---- batch-issue all 4 V fragment loads: in flight during softmax ---
        Frag16 vf[4];
        #pragma unroll
        for (int t = 0; t < 4; ++t) {
            const _Float16* vp = vbase + (size_t)(t * 16 + ln) * S_ + j0 + hi * 16;
            vf[t].h[0] = *(const v8h*)(vp);
            vf[t].h[1] = *(const v8h*)(vp + 8);
        }

        // ---- shared max: per-lane max of 16 values, then width-32 tree ------
        float tmax = fmaxf(s0[0], s1[0]);
        #pragma unroll
        for (int r = 1; r < 8; ++r) tmax = fmaxf(tmax, fmaxf(s0[r], s1[r]));
        #pragma unroll
        for (int off = 16; off >= 1; off >>= 1)
            tmax = fmaxf(tmax, __shfl_xor(tmax, off, 32));

        const float mnew = fmaxf(mrun, tmax);
        if (mnew > mrun) {                       // wave-uniform scalar branch
            const float alpha = fast_exp2(mrun - mnew);
            #pragma unroll
            for (int t = 0; t < 4; ++t)
                #pragma unroll
                for (int r = 0; r < 8; ++r) o[t][r] *= alpha;
            #pragma unroll
            for (int r = 0; r < 8; ++r) lacc[r] *= alpha;
            mrun = mnew;
        }

        // ---- P = exp2(s - m): stage C-layout f32 -> A-layout f16 via LDS ----
        #pragma unroll
        for (int r = 0; r < 8; ++r) {
            Pst[wave][r + 8 * hi][ln]      = (_Float16)fast_exp2(s0[r] - mrun);
            Pst[wave][r + 8 * hi][16 + ln] = (_Float16)fast_exp2(s1[r] - mrun);
        }
        Frag16 pf;   // in-wave LDS ops are in-order (DScnt); no barrier needed
        pf.h[0] = *(const v8h*)&Pst[wave][ln][hi * 16];
        pf.h[1] = *(const v8h*)&Pst[wave][ln][hi * 16 + 8];

        // ---- row sums via WMMA against all-ones B (independent of V loads) --
        lacc = wmma_f16(pf.v, ones.v, lacc);

        // ---- O += P @ V  (V fragments already resident) ---------------------
        #pragma unroll
        for (int t = 0; t < 4; ++t)
            o[t] = wmma_f16(pf.v, vf[t].v, o[t]);
    }

    // ---- normalize + store f32 (l replicated across lanes: no reduction) ---
    #pragma unroll
    for (int r = 0; r < 8; ++r) {
        const float inv = 1.0f / lacc[r];
        const int srow = m0 + r + 8 * hi;
        float* op = outp + ((size_t)(b * S_ + srow) * NH_ + h) * 64;
        #pragma unroll
        for (int t = 0; t < 4; ++t) op[t * 16 + ln] = o[t][r] * inv;
    }
}

// ---------------------------------------------------------------------------
extern "C" void kernel_launch(void* const* d_in, const int* in_sizes, int n_in,
                              void* d_out, int out_size, void* d_ws, size_t ws_size,
                              hipStream_t stream) {
    (void)in_sizes; (void)n_in; (void)out_size; (void)ws_size;
    const float* x  = (const float*)d_in[0];
    const float* Wq = (const float*)d_in[1];
    const float* bq = (const float*)d_in[2];
    const float* Wk = (const float*)d_in[3];
    const float* bk = (const float*)d_in[4];
    const float* Wv = (const float*)d_in[5];
    const float* bv = (const float*)d_in[6];
    float* out = (float*)d_out;

    // workspace: Qh 16MB | Kh 4MB | Vt 4MB  (f16)
    _Float16* Qh = (_Float16*)d_ws;
    _Float16* Kh = Qh + (size_t)B_ * S_ * DK_;
    _Float16* Vt = Kh + (size_t)B_ * S_ * DKV_;

    // Q pre-scaled by (1/sqrt(64)) * log2(e) so attention is exp2-domain.
    const float qscale = 0.125f * 1.44269504088896340736f;

    const dim3 blk(128);
    proj_kernel<DK_,  false><<<dim3((B_ * S_) / 64, DK_  / 128), blk, 0, stream>>>(x, Wq, bq, Qh, qscale);
    proj_kernel<DKV_, false><<<dim3((B_ * S_) / 64, DKV_ / 128), blk, 0, stream>>>(x, Wk, bk, Kh, 1.0f);
    proj_kernel<DKV_, true ><<<dim3((B_ * S_) / 64, DKV_ / 128), blk, 0, stream>>>(x, Wv, bv, Vt, 1.0f);
    attn_kernel<<<dim3(S_ / 64, B_ * NH_), blk, 0, stream>>>(Qh, Kh, Vt, out);
}
